// RPN_58858231824761
// MI455X (gfx1250) — matrix operand order repaired and checked by reading, not compiled, not run on hardware
//
#include <hip/hip_runtime.h>
#include <hip/hip_bf16.h>

typedef __attribute__((ext_vector_type(16))) _Float16 v16h;
typedef __attribute__((ext_vector_type(8)))  _Float16 v8h;
typedef __attribute__((ext_vector_type(8)))  float    v8f;

#define IN_CH   256
#define TILE_M  64
#define LDA     40                        // A tile row stride (f16): 32 + 8 pad
#define LDB     40                        // B tile row stride (f16)
#define LDT     264                       // activation tile row stride (f16)
#define W_ELEMS (9*256*256)
#define STAGE_BYTES ((64*LDA + 256*LDB) * 2)   // 25600 B per pipeline buffer

// ---------------------------------------------------------------------------
// Kernel 1: pack fp32 weights into f16, WMMA-B friendly layouts.
//   wPack[tap][o][c]  (K = c contiguous)   tap = r*3+s
//   hPack[o][c]       o: 0..2 obj, 3..14 delta, 15 zero pad
// ---------------------------------------------------------------------------
__global__ void rpn_pack_weights(const float* __restrict__ conv_w,
                                 const float* __restrict__ obj_w,
                                 const float* __restrict__ delta_w,
                                 _Float16* __restrict__ wPack,
                                 _Float16* __restrict__ hPack) {
    int idx = blockIdx.x * 256 + threadIdx.x;
    if (idx < W_ELEMS) {
        int c   = idx & 255;
        int o   = (idx >> 8) & 255;
        int tap = idx >> 16;
        int r = tap / 3, s = tap % 3;
        wPack[idx] = (_Float16)conv_w[((o * IN_CH + c) * 3 + r) * 3 + s];
    } else {
        int h = idx - W_ELEMS;
        if (h < 16 * 256) {
            int c = h & 255;
            int o = h >> 8;
            float v = 0.0f;
            if (o < 3)       v = obj_w[o * IN_CH + c];
            else if (o < 15) v = delta_w[(o - 3) * IN_CH + c];
            hPack[o * 256 + c] = (_Float16)v;
        }
    }
}

__device__ __forceinline__ v16h mk16(v8h lo, v8h hi) {
    v16h r;
#pragma unroll
    for (int i = 0; i < 8; ++i) { r[i] = lo[i]; r[i + 8] = hi[i]; }
    return r;
}

struct BStage { float4 b0, b1, b2, b3; };

// ---------------------------------------------------------------------------
// Kernel 2: fused implicit-GEMM 3x3 conv + ReLU + both 1x1 heads.
// One block = 64 pixels x 256 out-channels. 8 waves: 4 (M) x 2 (N).
// Pipeline per K-step (after the top barrier, buffer (step+1)&1 is free):
//   issue B loads (L2-hot) -> issue A loads (HBM) -> ds_store B (short wait)
//   -> 8 WMMAs (A latency hides here) -> ds_store A.
// Only the 4-VGPR A stage lives across the compute block -> no spills.
// ---------------------------------------------------------------------------
__global__ void __launch_bounds__(256, 1)
rpn_fused(const float* __restrict__ f0, const float* __restrict__ f1,
          const float* __restrict__ f2, const float* __restrict__ f3,
          const float* __restrict__ f4,
          const float* __restrict__ conv_b,
          const float* __restrict__ obj_b,
          const float* __restrict__ delta_b,
          const _Float16* __restrict__ wPack,
          const _Float16* __restrict__ hPack,
          float* __restrict__ out) {
    __shared__ __align__(16) unsigned char smem[2 * STAGE_BYTES];  // 51200 B
    _Float16* Ts = (_Float16*)smem;        // [64][LDT], reused after main loop

    // ---- block -> (level, image, tile); fully unrolled so constants fold ----
    int bb = blockIdx.x;
    int level = 4, H = 13, W = 21, img = 0, tile = 0;
    bool found = false;
#pragma unroll
    for (int l = 0; l < 5; ++l) {
        const int hh = (l == 0) ? 200 : (l == 1) ? 100 : (l == 2) ? 50 : (l == 3) ? 25 : 13;
        const int ww = (l == 0) ? 336 : (l == 1) ? 168 : (l == 2) ? 84 : (l == 3) ? 42 : 21;
        const int t  = (hh * ww + TILE_M - 1) >> 6;
        if (!found && bb < 2 * t) {
            found = true; level = l; H = hh; W = ww;
            img = (bb >= t); tile = bb - (bb >= t ? t : 0);
        }
        if (!found) bb -= 2 * t;
    }
    const int P = H * W;
    const float* Xl = (level == 0) ? f0 : (level == 1) ? f1 : (level == 2) ? f2
                    : (level == 3) ? f3 : f4;
    const float* X = Xl + (long)img * IN_CH * P;
    const int pix0 = tile * TILE_M;

    const int tid  = threadIdx.x;
    const int lane = tid & 31;
    const int ln   = lane & 15;
    const int hi   = lane >> 4;            // 0: lanes 0-15, 1: lanes 16-31
    const int wave = tid >> 5;
    const int mW   = wave & 3;             // M sub-tile (16 pixels)
    const int nW   = wave >> 2;            // N half (128 channels)

    // staging roles: thread -> (pixel mm, channel octet jj)
    const int mm = tid & 63;
    const int jj = tid >> 6;
    const int pg = pix0 + mm;
    const bool pixOk = (pg < P);
    const int py = pg / W;
    const int px = pg - py * W;
    const unsigned chanOff = (unsigned)(jj * 8) * (unsigned)P;  // 32-bit offsets

    // fragment addressing (wave32 16-bit A/B layouts)
    const int am    = mW * 16 + ln;
    const int ak0   = hi * 16;             // A: K 0-15 / 16-31
    const int bk0   = hi * 16;             // B: K 0-15 / 16-31
    const int nBase = nW * 128;

    __builtin_prefetch(hPack + tid * 16, 0, 1);   // warm head weights (L2)

    v8f acc[8];
#pragma unroll
    for (int nb = 0; nb < 8; ++nb) acc[nb] = (v8f)0.0f;

    // ---- pipeline helpers ----
    auto loadB = [&](int step, BStage& bR) {
        const int tap = step >> 3;
        const int kc  = step & 7;
        const float4* wsrc =
            (const float4*)(wPack + (unsigned)(tap * 65536 + tid * 256 + kc * 32));
        bR.b0 = wsrc[0];
        bR.b1 = wsrc[1];
        bR.b2 = wsrc[2];
        bR.b3 = wsrc[3];
    };

    auto loadA = [&](int step, v8h& aR) {
        const int tap = step >> 3;
        const int kc  = step & 7;
        const int r = tap / 3;
        const int s = tap - r * 3;
        const int ys = py + r - 1;
        const int xs = px + s - 1;
        const bool pv = pixOk && ((unsigned)ys < (unsigned)H) &&
                        ((unsigned)xs < (unsigned)W);
        const unsigned poff = pv ? (unsigned)(ys * W + xs) : 0u;  // clamped, in-bounds
        const unsigned base = poff + chanOff + (unsigned)(kc * 32) * (unsigned)P;
#pragma unroll
        for (int q = 0; q < 8; ++q) {
            const float raw = X[base + (unsigned)q * (unsigned)P];  // independent addrs
            const float v = pv ? raw : 0.0f;                        // mask via cndmask
            aR[q] = (_Float16)v;
        }
    };

    auto storeB = [&](int buf, const BStage& bR) {
        _Float16* Bb = (_Float16*)(smem + buf * STAGE_BYTES) + 64 * LDA;
        float4* bdst = (float4*)&Bb[tid * LDB];
        bdst[0] = bR.b0;
        bdst[1] = bR.b1;
        bdst[2] = bR.b2;
        bdst[3] = bR.b3;
    };

    auto storeA = [&](int buf, const v8h& aR) {
        _Float16* Ab = (_Float16*)(smem + buf * STAGE_BYTES);
        *(v8h*)&Ab[mm * LDA + jj * 8] = aR;
    };

    auto computeStep = [&](int buf) {
        const _Float16* Ab = (const _Float16*)(smem + buf * STAGE_BYTES);
        const _Float16* Bb = Ab + 64 * LDA;
        const v16h af = mk16(*(const v8h*)&Ab[am * LDA + ak0],
                             *(const v8h*)&Ab[am * LDA + ak0 + 8]);
#pragma unroll
        for (int nb = 0; nb < 8; ++nb) {
            const _Float16* bp = &Bb[(nBase + nb * 16 + ln) * LDB + bk0];
            const v16h bf = mk16(*(const v8h*)bp, *(const v8h*)(bp + 8));
            acc[nb] = __builtin_amdgcn_wmma_f32_16x16x32_f16(
                false, af, false, bf, (short)0, acc[nb], false, false);
        }
    };

    // ---- main K loop: 72 steps (9 taps x 8 chunks of 32 channels) ----
    {
        v8h a0; BStage b0;
        loadB(0, b0);
        loadA(0, a0);
        storeB(0, b0);
        storeA(0, a0);
    }
    for (int step = 0; step < 72; ++step) {
        __syncthreads();          // buffer (step&1) ready; buffer (step+1)&1 free
        v8h aN;
        if (step < 71) {
            BStage bN;
            loadB(step + 1, bN);              // L2-hot weights, issued first
            loadA(step + 1, aN);              // HBM loads, latency spans compute
            storeB((step + 1) & 1, bN);       // short wait: B done, A outstanding
        }
        computeStep(step & 1);                // 8 WMMAs; A latency hides here
        if (step < 71) storeA((step + 1) & 1, aN);
    }
    __syncthreads();

    // ---- epilogue 1: bias + ReLU, spill f16 activation tile to LDS ----
#pragma unroll
    for (int nb = 0; nb < 8; ++nb) {
        const int ch = nBase + nb * 16 + ln;
        const float bias = conv_b[ch];
#pragma unroll
        for (int i = 0; i < 8; ++i) {
            const int m = mW * 16 + i + 8 * hi;  // C/D layout: vgpr i, lane half
            float t = acc[nb][i] + bias;
            t = t > 0.0f ? t : 0.0f;
            Ts[m * LDT + ch] = (_Float16)t;
        }
    }
    __syncthreads();

    // ---- epilogue 2: 1x1 heads as one WMMA GEMM, M=64 N=16 K=256 ----
    if (wave < 4) {
        v8f acc2 = (v8f)0.0f;
        const int am2 = wave * 16 + ln;
#pragma unroll
        for (int k0 = 0; k0 < 256; k0 += 32) {
            const v16h af = mk16(*(const v8h*)&Ts[am2 * LDT + k0 + ak0],
                                 *(const v8h*)&Ts[am2 * LDT + k0 + ak0 + 8]);
            const _Float16* hp = hPack + ln * 256 + k0 + bk0;
            const v16h bf = mk16(*(const v8h*)hp, *(const v8h*)(hp + 8));
            acc2 = __builtin_amdgcn_wmma_f32_16x16x32_f16(
                false, af, false, bf, (short)0, acc2, false, false);
        }
        const long logitOff = (level == 0) ? 0L      : (level == 1) ? 403200L
                            : (level == 2) ? 504000L : (level == 3) ? 529200L : 535500L;
        const long deltaOff = (level == 0) ? 537138L : (level == 1) ? 2149938L
                            : (level == 2) ? 2553138L: (level == 3) ? 2653938L : 2679138L;
        const int o = ln;
        const float hb = (o < 3) ? obj_b[o] : ((o < 15) ? delta_b[o - 3] : 0.0f);
#pragma unroll
        for (int i = 0; i < 8; ++i) {
            const int m = wave * 16 + i + 8 * hi;
            const int pgo = pix0 + m;
            if (pgo < P && o < 15) {
                const float v = acc2[i] + hb;
                long idx;
                if (o < 3)
                    idx = logitOff + (long)img * 3 * P + (long)o * P + pgo;
                else
                    idx = deltaOff + (long)img * 12 * P + (long)(o - 3) * P + pgo;
                out[idx] = v;
            }
        }
    }
}

// ---------------------------------------------------------------------------
extern "C" void kernel_launch(void* const* d_in, const int* in_sizes, int n_in,
                              void* d_out, int out_size, void* d_ws, size_t ws_size,
                              hipStream_t stream) {
    const float* f0      = (const float*)d_in[0];
    const float* f1      = (const float*)d_in[1];
    const float* f2      = (const float*)d_in[2];
    const float* f3      = (const float*)d_in[3];
    const float* f4      = (const float*)d_in[4];
    const float* conv_w  = (const float*)d_in[5];
    const float* conv_b  = (const float*)d_in[6];
    const float* obj_w   = (const float*)d_in[7];
    const float* obj_b   = (const float*)d_in[8];
    const float* delta_w = (const float*)d_in[9];
    const float* delta_b = (const float*)d_in[10];

    _Float16* wPack = (_Float16*)d_ws;            // 9*256*256 f16 = 1.13 MB
    _Float16* hPack = wPack + W_ELEMS;            // 16*256 f16

    const int packThreads = W_ELEMS + 16 * 256;
    rpn_pack_weights<<<(packThreads + 255) / 256, 256, 0, stream>>>(
        conv_w, obj_w, delta_w, wPack, hPack);

    // total tiles: sum over levels of 2 * ceil(H*W / 64)
    const int Hs[5] = {200, 100, 50, 25, 13};
    const int Ws5[5] = {336, 168, 84, 42, 21};
    int blocks = 0;
    for (int l = 0; l < 5; ++l) blocks += 2 * ((Hs[l] * Ws5[l] + 63) / 64);

    rpn_fused<<<blocks, 256, 0, stream>>>(f0, f1, f2, f3, f4,
                                          conv_b, obj_b, delta_b,
                                          wPack, hPack, (float*)d_out);
}
